// MultiheadSelfAttention_55808805044472
// MI455X (gfx1250) — compile-verified, matrix operand order checked by reading, not run on hardware
//
#include <hip/hip_runtime.h>

// MI455X / gfx1250 multi-head self-attention with RoPE.
//   B=2, S=2048, D=1024, H=16, dh=64.
// Pipeline: (1) fused QKV projection + RoPE -> f16 workspace
//           (2) flash-attention (online softmax, causal) -> f16 ctx
//           (3) output projection -> f32 out
// GEMMs use v_wmma_f32_16x16x32_f16. Projection passes stage the shared
// 64x32 weight slab per block in LDS (f16, fragment-ready, bank-padded),
// via the CDNA5 async global->LDS copies (ASYNCcnt) when available.

typedef __attribute__((ext_vector_type(16))) _Float16 v16h;
typedef __attribute__((ext_vector_type(8)))  _Float16 v8h;
typedef __attribute__((ext_vector_type(8)))  float    v8f;
typedef __attribute__((ext_vector_type(4)))  float    v4f;
typedef __attribute__((ext_vector_type(4)))  int      v4i;

#define S_LEN 2048
#define D_DIM 1024
#define H_NUM 16
#define DH    64
#define B_NUM 2
#define M_ROWS (B_NUM * S_LEN)   // 4096

// Padded LDS row stride for the weight slab: 40 halves = 80B so the 16 lanes
// of a fragment read map to 16 distinct banks (20*l mod 64 distinct, l<16),
// while keeping 16B alignment for b128 ds loads.
#define WSLAB_LD 40

// ---------------------------------------------------------------------------
// CDNA5 async global->LDS copy (GLOBAL_LOAD_ASYNC_TO_LDS_B128, ASYNCcnt).
// Probe-confirmed signature: (v4i addrspace(1)*, v4i addrspace(3)*, imm, imm).
// ---------------------------------------------------------------------------
#if defined(__has_builtin)
#if __has_builtin(__builtin_amdgcn_global_load_async_to_lds_b128)
#define ASYNC_LDS 1
#endif
#endif
#ifndef ASYNC_LDS
#define ASYNC_LDS 0
#endif

#if ASYNC_LDS
typedef __attribute__((address_space(1))) v4i gv4i;
typedef __attribute__((address_space(3))) v4i lv4i;
__device__ __forceinline__ void async_copy_b128(const float* src, float* dstLds) {
  __builtin_amdgcn_global_load_async_to_lds_b128((gv4i*)src, (lv4i*)dstLds, 0, 0);
}
#endif

__device__ __forceinline__ void wait_async0() {
#if defined(__has_builtin)
#if __has_builtin(__builtin_amdgcn_s_wait_asynccnt)
  __builtin_amdgcn_s_wait_asynccnt(0);
#else
  asm volatile("s_wait_asynccnt 0x0" ::: "memory");
#endif
#else
  asm volatile("s_wait_asynccnt 0x0" ::: "memory");
#endif
}

__device__ __forceinline__ v8f zero8() {
  v8f r;
#pragma unroll
  for (int i = 0; i < 8; ++i) r[i] = 0.0f;
  return r;
}

__device__ __forceinline__ v8f wmma16x16x32(v16h a, v16h b, v8f c) {
  // 8 args: (neg_a, A, neg_b, B, c_mod, C, reuse_a, reuse_b)
  return __builtin_amdgcn_wmma_f32_16x16x32_f16(false, a, false, b, (short)0, c,
                                                false, false);
}

// Per-lane share of a 16x32 f16 fragment from an f32 row-major matrix with
// on-the-fly f16 conversion. Lane holds K chunks [cb..cb+7], [cb+16..cb+23]
// of its row (cb = (lane>>4)*8).
__device__ __forceinline__ v16h load_frag_cvt(const float* __restrict__ base,
                                              int row, int cb, int ld) {
  const float* p = base + (size_t)row * ld + cb;
  v4f a0 = *(const v4f*)(p);
  v4f a1 = *(const v4f*)(p + 4);
  v4f b0 = *(const v4f*)(p + 16);
  v4f b1 = *(const v4f*)(p + 20);
  v16h r;
#pragma unroll
  for (int i = 0; i < 4; ++i) {
    r[i]      = (_Float16)a0[i];
    r[4 + i]  = (_Float16)a1[i];
    r[8 + i]  = (_Float16)b0[i];
    r[12 + i] = (_Float16)b1[i];
  }
  return r;
}

// Same but from an f16 matrix (two contiguous b128 loads).
__device__ __forceinline__ v16h load_frag_h(const _Float16* __restrict__ base,
                                            int row, int cb, int ld) {
  const _Float16* p = base + (size_t)row * ld + cb;
  v8h lo = *(const v8h*)(p);
  v8h hi = *(const v8h*)(p + 16);
  v16h r;
#pragma unroll
  for (int i = 0; i < 8; ++i) { r[i] = lo[i]; r[8 + i] = hi[i]; }
  return r;
}

// Fragment read from the LDS weight slab (fragment-ready f16 layout).
__device__ __forceinline__ v16h load_frag_slab(const _Float16* slab, int row,
                                               int cb) {
  const _Float16* p = slab + row * WSLAB_LD + cb;
  v8h lo = *(const v8h*)p;
  v8h hi = *(const v8h*)(p + 16);
  v16h r;
#pragma unroll
  for (int i = 0; i < 8; ++i) { r[i] = lo[i]; r[8 + i] = hi[i]; }
  return r;
}

// ---------------------------------------------------------------------------
// Pass 1: Q/K/V = x @ W{q,k,v}^T, RoPE on Q and K.
// One block (8 waves) computes a 128x64 tile of one projection; the 64x32
// weight k-slab is staged once per block in LDS (async path when available).
// Tiles: mt128 in [0,32), nt in [0,16), mat in {0,1,2} -> 1536 blocks.
// Q,K stored f16 [bh][s][dh]; V stored f16 transposed [bh][dh][s].
// ---------------------------------------------------------------------------
__global__ __launch_bounds__(256) void qkv_rope_kernel(
    const float* __restrict__ x, const int* __restrict__ positions,
    const float* __restrict__ Wq, const float* __restrict__ Wk,
    const float* __restrict__ Wv,
    _Float16* __restrict__ Qws, _Float16* __restrict__ Kws,
    _Float16* __restrict__ Vt) {
  __shared__ __align__(16) _Float16 wslab[64 * WSLAB_LD];
#if ASYNC_LDS
  __shared__ __align__(16) float wstage[64 * 32];
#endif
  int tid  = threadIdx.x;
  int wave = tid >> 5;
  int lane = tid & 31;
  int blk   = blockIdx.x;
  int mt128 = blk & 31;
  int nt    = (blk >> 5) & 15;
  int mat   = blk >> 9;
  const float* W = (mat == 0) ? Wq : (mat == 1) ? Wk : Wv;

  int half = lane >> 4;
  int lidx = lane & 15;
  int cb   = half * 8;
  int arow = mt128 * 128 + wave * 16 + lidx;

  int sr = tid >> 2;          // staging row 0..63
  int sc = (tid & 3) * 8;     // staging col 0,8,16,24
  const float* wsrc = W + (size_t)(nt * 64 + sr) * D_DIM + sc;

  v8f acc[4];
#pragma unroll
  for (int j = 0; j < 4; ++j) acc[j] = zero8();

  for (int kk = 0; kk < D_DIM; kk += 32) {
    // ---- stage W[nt*64 .. +63][kk..kk+31] as f16 into wslab ----
#if ASYNC_LDS
    async_copy_b128(wsrc + kk,     &wstage[sr * 32 + sc]);
    async_copy_b128(wsrc + kk + 4, &wstage[sr * 32 + sc + 4]);
    wait_async0();
    __syncthreads();
    v4f f0 = *(const v4f*)&wstage[sr * 32 + sc];
    v4f f1 = *(const v4f*)&wstage[sr * 32 + sc + 4];
#else
    v4f f0 = *(const v4f*)(wsrc + kk);
    v4f f1 = *(const v4f*)(wsrc + kk + 4);
#endif
    v8h hh;
#pragma unroll
    for (int i = 0; i < 4; ++i) {
      hh[i]     = (_Float16)f0[i];
      hh[4 + i] = (_Float16)f1[i];
    }
    *(v8h*)&wslab[sr * WSLAB_LD + sc] = hh;
    __syncthreads();

    // ---- compute: each wave does 4 WMMAs against the shared slab ----
    v16h a = load_frag_cvt(x, arow, kk + cb, D_DIM);
#pragma unroll
    for (int j = 0; j < 4; ++j) {
      v16h b = load_frag_slab(wslab, j * 16 + lidx, cb);
      acc[j] = wmma16x16x32(a, b, acc[j]);
    }
    __syncthreads();   // protect slab before next iteration overwrites it
  }

  // Epilogue: RoPE (Q,K only) + store. C-fragment: lane holds col
  // n = nt*64 + j*16 + lidx, rows m = v + 8*half. Even/odd head dims are in
  // adjacent lanes -> pair exchange via shfl_xor(1).
  int mt16 = mt128 * 8 + wave;
#pragma unroll
  for (int j = 0; j < 4; ++j) {
    int n = nt * 64 + j * 16 + lidx;
    int h = n >> 6;
    int d = n & 63;
    float invf = __expf(-((float)(d & ~1) * (1.0f / 64.0f)) * 9.2103403719761836f);
#pragma unroll
    for (int v = 0; v < 8; ++v) {
      int m = mt16 * 16 + v + half * 8;
      int b = m >> 11;          // /S_LEN
      int s = m & (S_LEN - 1);
      float val   = acc[j][v];
      float other = __shfl_xor(val, 1, 32);
      float res;
      if (mat < 2) {
        float ang = (float)positions[s] * invf;
        float cs  = __cosf(ang);
        float sn  = __sinf(ang);
        res = (d & 1) ? (other * sn + val * cs) : (val * cs - other * sn);
      } else {
        res = val;
      }
      size_t bh = (size_t)(b * H_NUM + h);
      if (mat == 0)
        Qws[(bh * S_LEN + s) * DH + d] = (_Float16)res;
      else if (mat == 1)
        Kws[(bh * S_LEN + s) * DH + d] = (_Float16)res;
      else
        Vt[(bh * DH + d) * S_LEN + s] = (_Float16)res;
    }
  }
}

// ---------------------------------------------------------------------------
// Pass 2: flash attention. One wave owns one 16-query tile of one (b,h).
// Streams keys in chunks of 32; online softmax with causal mask; P staged
// through a per-wave LDS tile to transpose into A-fragment layout for P x V.
// ---------------------------------------------------------------------------
__global__ __launch_bounds__(256) void attn_kernel(
    const _Float16* __restrict__ Q, const _Float16* __restrict__ K,
    const _Float16* __restrict__ Vt, _Float16* __restrict__ Ctx) {
  __shared__ __align__(16) _Float16 plds[8][16 * 32];

  int wave = threadIdx.x >> 5;
  int lane = threadIdx.x & 31;
  int tile = blockIdx.x * 8 + wave;
  int qt = tile & 127;          // query tile within (b,h)
  int bh = tile >> 7;
  int b  = bh >> 4;
  int h  = bh & 15;

  const _Float16* Qh = Q  + (size_t)bh * S_LEN * DH;
  const _Float16* Kh = K  + (size_t)bh * S_LEN * DH;
  const _Float16* Vh = Vt + (size_t)bh * DH * S_LEN;

  int half = lane >> 4;
  int lidx = lane & 15;
  int cb   = half * 8;

  // Preload Q A-fragments for the two dh chunks (K-dim = dh = 64).
  v16h qa[2];
#pragma unroll
  for (int c = 0; c < 2; ++c)
    qa[c] = load_frag_h(Qh, qt * 16 + lidx, c * 32 + cb, DH);

  float rmax[8], rsum[8];
#pragma unroll
  for (int v = 0; v < 8; ++v) { rmax[v] = -3.0e38f; rsum[v] = 0.0f; }
  v8f o[4];
#pragma unroll
  for (int j = 0; j < 4; ++j) o[j] = zero8();

  _Float16* myP = &plds[wave][0];
  const float scale = 0.125f;   // 1/sqrt(64)

  for (int kk = 0; kk < (qt + 1) * 16; kk += 32) {
    // scores for 32 keys: two 16-wide key subtiles x two dh chunks
    v8f s0 = zero8(), s1 = zero8();
#pragma unroll
    for (int c = 0; c < 2; ++c) {
      v16h kb0 = load_frag_h(Kh, kk + lidx, c * 32 + cb, DH);
      s0 = wmma16x16x32(qa[c], kb0, s0);
      v16h kb1 = load_frag_h(Kh, kk + 16 + lidx, c * 32 + cb, DH);
      s1 = wmma16x16x32(qa[c], kb1, s1);
    }

    // Online softmax update (row m = v + 8*half lives on one 16-lane half).
#pragma unroll
    for (int v = 0; v < 8; ++v) {
      int q  = qt * 16 + v + half * 8;
      int k0 = kk + lidx;
      int k1 = k0 + 16;
      float a0 = (k0 <= q) ? s0[v] * scale : -3.0e38f;
      float a1 = (k1 <= q) ? s1[v] * scale : -3.0e38f;
      float mx = fmaxf(a0, a1);
#pragma unroll
      for (int off = 1; off < 16; off <<= 1)
        mx = fmaxf(mx, __shfl_xor(mx, off, 32));
      float mnew  = fmaxf(rmax[v], mx);
      float alpha = __expf(rmax[v] - mnew);
      rmax[v] = mnew;
      float p0 = __expf(a0 - mnew);
      float p1 = __expf(a1 - mnew);
      float ps = p0 + p1;
#pragma unroll
      for (int off = 1; off < 16; off <<= 1)
        ps += __shfl_xor(ps, off, 32);
      rsum[v] = rsum[v] * alpha + ps;
#pragma unroll
      for (int j = 0; j < 4; ++j) o[j][v] *= alpha;
      int prow = v + half * 8;
      myP[prow * 32 + lidx]      = (_Float16)p0;
      myP[prow * 32 + 16 + lidx] = (_Float16)p1;
    }

    // Same-wave LDS RAW: force store completion (CDNA5 split DS counter).
    asm volatile("s_wait_dscnt 0" ::: "memory");

    // Re-read P in A-fragment layout (lane = query row, K chunks of key dim).
    v16h pa;
    {
      const _Float16* pp = myP + lidx * 32 + cb;
      v8h lo = *(const v8h*)(pp);
      v8h hi = *(const v8h*)(pp + 16);
#pragma unroll
      for (int i = 0; i < 8; ++i) { pa[i] = lo[i]; pa[8 + i] = hi[i]; }
    }

    // ctx += P x V  (B-fragment: lane = output dim d, contiguous along keys
    // thanks to the transposed V layout).
#pragma unroll
    for (int j = 0; j < 4; ++j) {
      v16h vb = load_frag_h(Vh, j * 16 + lidx, kk + cb, S_LEN);
      o[j] = wmma16x16x32(pa, vb, o[j]);
    }
  }

  // Normalize and store ctx as f16 [B*S][D] for the output projection.
#pragma unroll
  for (int v = 0; v < 8; ++v) {
    float inv = 1.0f / rsum[v];
    int s = qt * 16 + v + half * 8;
    size_t row = (size_t)b * S_LEN + s;
#pragma unroll
    for (int j = 0; j < 4; ++j) {
      int d = j * 16 + lidx;
      Ctx[row * D_DIM + h * DH + d] = (_Float16)(o[j][v] * inv);
    }
  }
}

// ---------------------------------------------------------------------------
// Pass 3: out = ctx @ Wo^T (f32 output). One block per 128x64 tile with the
// same LDS weight-slab staging as pass 1. 32*16 = 512 blocks.
// ---------------------------------------------------------------------------
__global__ __launch_bounds__(256) void out_proj_kernel(
    const _Float16* __restrict__ Ctx, const float* __restrict__ Wo,
    float* __restrict__ out) {
  __shared__ __align__(16) _Float16 wslab[64 * WSLAB_LD];
#if ASYNC_LDS
  __shared__ __align__(16) float wstage[64 * 32];
#endif
  int tid  = threadIdx.x;
  int wave = tid >> 5;
  int lane = tid & 31;
  int blk   = blockIdx.x;
  int mt128 = blk & 31;
  int nt    = blk >> 5;

  int half = lane >> 4;
  int lidx = lane & 15;
  int cb   = half * 8;
  int arow = mt128 * 128 + wave * 16 + lidx;

  int sr = tid >> 2;
  int sc = (tid & 3) * 8;
  const float* wsrc = Wo + (size_t)(nt * 64 + sr) * D_DIM + sc;

  v8f acc[4];
#pragma unroll
  for (int j = 0; j < 4; ++j) acc[j] = zero8();

  for (int kk = 0; kk < D_DIM; kk += 32) {
#if ASYNC_LDS
    async_copy_b128(wsrc + kk,     &wstage[sr * 32 + sc]);
    async_copy_b128(wsrc + kk + 4, &wstage[sr * 32 + sc + 4]);
    wait_async0();
    __syncthreads();
    v4f f0 = *(const v4f*)&wstage[sr * 32 + sc];
    v4f f1 = *(const v4f*)&wstage[sr * 32 + sc + 4];
#else
    v4f f0 = *(const v4f*)(wsrc + kk);
    v4f f1 = *(const v4f*)(wsrc + kk + 4);
#endif
    v8h hh;
#pragma unroll
    for (int i = 0; i < 4; ++i) {
      hh[i]     = (_Float16)f0[i];
      hh[4 + i] = (_Float16)f1[i];
    }
    *(v8h*)&wslab[sr * WSLAB_LD + sc] = hh;
    __syncthreads();

    v16h a = load_frag_h(Ctx, arow, kk + cb, D_DIM);
#pragma unroll
    for (int j = 0; j < 4; ++j) {
      v16h b = load_frag_slab(wslab, j * 16 + lidx, cb);
      acc[j] = wmma16x16x32(a, b, acc[j]);
    }
    __syncthreads();
  }

  int mt16 = mt128 * 8 + wave;
#pragma unroll
  for (int v = 0; v < 8; ++v) {
    int m = mt16 * 16 + v + half * 8;
#pragma unroll
    for (int j = 0; j < 4; ++j)
      out[(size_t)m * D_DIM + nt * 64 + j * 16 + lidx] = acc[j][v];
  }
}

// ---------------------------------------------------------------------------
extern "C" void kernel_launch(void* const* d_in, const int* in_sizes, int n_in,
                              void* d_out, int out_size, void* d_ws,
                              size_t ws_size, hipStream_t stream) {
  (void)in_sizes; (void)n_in; (void)out_size; (void)ws_size;
  const float* x         = (const float*)d_in[0];
  const int*   positions = (const int*)d_in[1];
  const float* Wq = (const float*)d_in[2];
  const float* Wk = (const float*)d_in[3];
  const float* Wv = (const float*)d_in[4];
  const float* Wo = (const float*)d_in[5];
  float* out = (float*)d_out;

  // Workspace: 4 x 8MB f16 buffers (Q, K, V^T, ctx) = 32 MB.
  const size_t NELT = (size_t)M_ROWS * D_DIM;   // 4,194,304
  _Float16* Qws = (_Float16*)d_ws;
  _Float16* Kws = Qws + NELT;
  _Float16* Vt  = Kws + NELT;
  _Float16* Ctx = Vt + NELT;

  // Pass 1: 3 mats * 16 ntiles * 32 mtiles(128-row) = 1536 blocks.
  qkv_rope_kernel<<<1536, 256, 0, stream>>>(x, positions, Wq, Wk, Wv,
                                            Qws, Kws, Vt);
  // Pass 2: 2*16*128 = 4096 q-tiles / 8 waves per block.
  attn_kernel<<<512, 256, 0, stream>>>(Qws, Kws, Vt, Ctx);
  // Pass 3: 16 ntiles * 32 mtiles = 512 blocks.
  out_proj_kernel<<<512, 256, 0, stream>>>(Ctx, Wo, out);
}
